// BiMamba_31971736552177
// MI455X (gfx1250) — matrix operand order repaired and checked by reading, not compiled, not run on hardware
//
#include <hip/hip_runtime.h>
#include <hip/hip_bf16.h>
#include <math.h>

// ---------------- problem constants ----------------
#define DMODEL 1024
#define DSTATE 16
#define DCONV  4
#define DINNER 2048
#define DTRANK 64
#define XDBLN  96          // DTRANK + 2*DSTATE
#define NB     2
#define SEQL   2048
#define MROWS  (NB*SEQL)   // 4096
#define NCHUNK 16
#define CHUNK  128         // SEQL / NCHUNK

typedef __attribute__((ext_vector_type(16))) _Float16 v16h;
typedef __attribute__((ext_vector_type(8)))  _Float16 v8h;
typedef __attribute__((ext_vector_type(8)))  float    v8f;
typedef int v4i __attribute__((vector_size(16)));    // matches builtin param type

// ---------------------------------------------------------------------
// async global->LDS copy (16B per lane), double buffered.  Builtin param
// types (from hipcc diagnostics): (v4i AS1* src, v4i AS3* dst, imm, imm).
// ---------------------------------------------------------------------
#if __has_builtin(__builtin_amdgcn_global_load_async_to_lds_b128)
#define ASYNC_CP(dst, src)                                                   \
    __builtin_amdgcn_global_load_async_to_lds_b128(                          \
        (__attribute__((address_space(1))) v4i*)(src),                       \
        (__attribute__((address_space(3))) v4i*)(dst), 0, 0)
#define WAIT_ASYNC() asm volatile("s_wait_asynccnt 0" ::: "memory")
#else
#define ASYNC_CP(dst, src) (*(v8h*)(dst) = *(const v8h*)(src))
#define WAIT_ASYNC()
#endif

#define LA_HALF (32 * 32)   // one A buffer: 32 rows x 32 k
#define LB_HALF (64 * 32)   // one B buffer: 64 cols x 32 k

// issue one 32-wide k-slab into LDS buffer p (block tile 32M x 64N)
__device__ __forceinline__ void gemm_issue(const _Float16* __restrict__ A,
                                           const _Float16* __restrict__ W,
                                           int K, int N, int kk, int p,
                                           _Float16* lA, _Float16* lB)
{
    const int tid = threadIdx.x;
    const int m0  = blockIdx.y * 32;
    const int n0  = blockIdx.x * 64;
    {   // B tile: 64 cols x 32 halves (all 256 threads, 16B each)
        int c = tid >> 2, seg = tid & 3;
        int cg = n0 + c; if (cg >= N) cg = N - 1;           // clamp edge (N=96)
        ASYNC_CP(lB + p * LB_HALF + c * 32 + seg * 8,
                 W + (size_t)cg * K + kk + seg * 8);
    }
    if (tid < 128) {  // A tile: 32 rows x 32 halves
        int r = tid >> 2, seg = tid & 3;
        ASYNC_CP(lA + p * LA_HALF + r * 32 + seg * 8,
                 A + (size_t)(m0 + r) * K + kk + seg * 8);
    }
}

// =====================================================================
// Block-cooperative WMMA GEMM core.  Block = 256 thr = 8 waves in a
// 2(M) x 4(N) tile grid -> 32x64 output per block.  K stepped by 32 with
// double-buffered async LDS staging.  Fragment layouts per ISA 7.12.2:
//   A 16x32 f16: lane<16 -> row=lane,   halves K {0..7,16..23}
//                lane>=16-> row=lane-16,halves K {8..15,24..31}
//   B 32x16 f16: lane<16 -> col=lane, K {0..7,16..23}; lane>=16 -> K {8..15,24..31}
// =====================================================================
__device__ __forceinline__ v8f gemm_block(const _Float16* __restrict__ A,
                                          const _Float16* __restrict__ W,
                                          int K, int N,
                                          _Float16* lA, _Float16* lB)
{
    const int lane = threadIdx.x & 31;
    const int wid  = threadIdx.x >> 5;
    const int hi   = lane >> 4;
    const int arow = (wid >> 2) * 16 + (lane & 15);
    const int bcol = (wid &  3) * 16 + (lane & 15);
    const int akb  = hi ? 8 : 0;
    const int bkb  = hi ? 8 : 0;

    v8f acc = {};
    const int nk = K >> 5;
    gemm_issue(A, W, K, N, 0, 0, lA, lB);
    for (int i = 0; i < nk; ++i) {
        WAIT_ASYNC();          // this wave's outstanding slab has landed
        __syncthreads();       // all waves' slabs visible; prev compute done
        if (i + 1 < nk)        // overlap next fill with this wmma step
            gemm_issue(A, W, K, N, (i + 1) << 5, (i + 1) & 1, lA, lB);

        const _Float16* la = lA + (i & 1) * LA_HALF + arow * 32 + akb;
        const _Float16* lb = lB + (i & 1) * LB_HALF + bcol * 32 + bkb;
        v8h a0 = *(const v8h*)la;        // ds_load_b128
        v8h a1 = *(const v8h*)(la + 16);
        v8h b0 = *(const v8h*)lb;
        v8h b1 = *(const v8h*)(lb + 16);
        v16h af, bf;
#pragma unroll
        for (int j = 0; j < 8; ++j) { af[j] = a0[j]; af[j + 8] = a1[j]; }
#pragma unroll
        for (int j = 0; j < 8; ++j) { bf[j] = b0[j]; bf[j + 8] = b1[j]; }
        acc = __builtin_amdgcn_wmma_f32_16x16x32_f16(false, af, false, bf,
                                                     (short)0, acc, false, false);
    }
    return acc;
}

// C/D layout: VGPR v, lane l : row = 16*wm + v + (l>>4)*8 ; col = 16*wn + (l&15)
#define TILE_COORDS()                                                        \
    const int lane  = threadIdx.x & 31;                                      \
    const int wid   = threadIdx.x >> 5;                                      \
    const int rbase = blockIdx.y * 32 + (wid >> 2) * 16 + (lane >> 4) * 8;   \
    const int col   = blockIdx.x * 64 + (wid & 3) * 16 + (lane & 15);

#define GEMM_SHARED()                 \
    __shared__ _Float16 lA[2 * LA_HALF]; \
    __shared__ _Float16 lB[2 * LB_HALF];

// ---------- GEMM 1: xz = x16 @ in_w16^T, split into xi16 / z16 ----------
__global__ void k_gemm_split(const _Float16* __restrict__ A, const _Float16* __restrict__ W,
                             _Float16* __restrict__ o_xi, _Float16* __restrict__ o_z,
                             int K, int N)
{
    GEMM_SHARED(); TILE_COORDS();
    v8f acc = gemm_block(A, W, K, N, lA, lB);
#pragma unroll
    for (int v = 0; v < 8; ++v) {
        int row = rbase + v;
        if (col < DINNER) o_xi[(size_t)row * DINNER + col]            = (_Float16)acc[v];
        else              o_z [(size_t)row * DINNER + (col - DINNER)] = (_Float16)acc[v];
    }
}

// ---------- GEMM 2: x_dbl = xc16 @ xp_w16^T  (N=96), also dt16 ----------
__global__ void k_gemm_xdbl(const _Float16* __restrict__ A, const _Float16* __restrict__ W,
                            float* __restrict__ xdbl, _Float16* __restrict__ dt16,
                            int K, int N)
{
    GEMM_SHARED(); TILE_COORDS();
    v8f acc = gemm_block(A, W, K, N, lA, lB);
    if (col < N) {
#pragma unroll
        for (int v = 0; v < 8; ++v) {
            int row = rbase + v;
            xdbl[(size_t)row * XDBLN + col] = acc[v];
            if (col < DTRANK) dt16[(size_t)row * DTRANK + col] = (_Float16)acc[v];
        }
    }
}

// ---------- GEMM 3: delta = softplus(dt16 @ dtp_w16^T + dtp_b) ----------
__global__ void k_gemm_delta(const _Float16* __restrict__ A, const _Float16* __restrict__ W,
                             const float* __restrict__ bias, float* __restrict__ delta,
                             int K, int N)
{
    GEMM_SHARED(); TILE_COORDS();
    v8f acc = gemm_block(A, W, K, N, lA, lB);
    float bv = bias[col];
#pragma unroll
    for (int v = 0; v < 8; ++v) {
        int row = rbase + v;
        float d = acc[v] + bv;
        d = (d > 20.f) ? d : logf(1.f + __expf(d));      // softplus
        delta[(size_t)row * DINNER + col] = d;
    }
}

// ---------- GEMM 4: out (+)= yact16 @ out_w16^T ----------
__global__ void k_gemm_out(const _Float16* __restrict__ A, const _Float16* __restrict__ W,
                           float* __restrict__ out, int K, int N, int accum)
{
    GEMM_SHARED(); TILE_COORDS();
    v8f acc = gemm_block(A, W, K, N, lA, lB);
#pragma unroll
    for (int v = 0; v < 8; ++v) {
        int row = rbase + v;
        size_t o = (size_t)row * N + col;
        out[o] = accum ? (out[o] + acc[v]) : acc[v];
    }
}

// =====================================================================
// elementwise / conversion kernels
// =====================================================================
__global__ void k_cvt_f32_f16(const float* __restrict__ src, _Float16* __restrict__ dst, int n)
{
    int i = blockIdx.x * 256 + threadIdx.x;
    if (i < n) dst[i] = (_Float16)src[i];
}

// x16[b,l,:] = (f16) x[b, flip? L-1-l : l, :]
__global__ void k_flip_cvt_x(const float* __restrict__ x, _Float16* __restrict__ x16, int flip)
{
    int i = blockIdx.x * 256 + threadIdx.x;
    if (i >= MROWS * DMODEL) return;
    int m = i / DMODEL, c = i - m * DMODEL;
    int b = m / SEQL,   l = m - b * SEQL;
    int ls = flip ? (SEQL - 1 - l) : l;
    x16[i] = (_Float16)x[((size_t)b * SEQL + ls) * DMODEL + c];
}

// causal depthwise conv(K=4) + bias + silu ; writes f32 and f16 copies
__global__ void k_conv_silu(const _Float16* __restrict__ xi, const float* __restrict__ cw,
                            const float* __restrict__ cb, float* __restrict__ xc32,
                            _Float16* __restrict__ xc16)
{
    int i = blockIdx.x * 256 + threadIdx.x;
    if (i >= MROWS * DINNER) return;
    int m = i / DINNER, d = i - m * DINNER;
    int b = m / SEQL,   l = m - b * SEQL;
    float s = cb[d];
#pragma unroll
    for (int k = 0; k < DCONV; ++k) {
        int ls = l - (DCONV - 1) + k;
        if (ls >= 0) s += cw[d * DCONV + k] * (float)xi[((size_t)b * SEQL + ls) * DINNER + d];
    }
    s = s / (1.f + __expf(-s));                          // silu
    xc32[i] = s;
    xc16[i] = (_Float16)s;
}

// =====================================================================
// chunked selective scan (3 passes)
// =====================================================================
__global__ void k_scan1(const float* __restrict__ delta, const float* __restrict__ xc32,
                        const float* __restrict__ xdbl,  const float* __restrict__ A_log,
                        float* __restrict__ ybuf, float* __restrict__ sdelta,
                        float* __restrict__ Q)
{
    __shared__ float lB[CHUNK][DSTATE];
    __shared__ float lC[CHUNK][DSTATE];
    int tmp = blockIdx.x;
    int dblk = tmp & 7; int c = (tmp >> 3) & 15; int b = tmp >> 7;
    int d0 = dblk * 256;

    for (int i = threadIdx.x; i < CHUNK * 32; i += 256) {
        int t = i >> 5, j = i & 31;
        float v = xdbl[((size_t)(b * SEQL + c * CHUNK + t)) * XDBLN + DTRANK + j];
        if (j < DSTATE) lB[t][j] = v; else lC[t][j - DSTATE] = v;
    }
    __syncthreads();

    int d = d0 + threadIdx.x;
    float a[DSTATE], h[DSTATE];
#pragma unroll
    for (int n = 0; n < DSTATE; ++n) { a[n] = -__expf(A_log[d * DSTATE + n]); h[n] = 0.f; }

    float sd = 0.f;
    for (int t = 0; t < CHUNK; ++t) {
        size_t row = (size_t)b * SEQL + c * CHUNK + t;
        float de = delta[row * DINNER + d];
        float u  = xc32 [row * DINNER + d];
        float du = de * u;
        sd += de;
        float y = 0.f;
#pragma unroll
        for (int n = 0; n < DSTATE; ++n) {
            h[n] = h[n] * __expf(de * a[n]) + du * lB[t][n];
            y += h[n] * lC[t][n];
        }
        ybuf[row * DINNER + d] = y;
    }
    size_t qb = ((size_t)(b * NCHUNK + c) * DINNER + d);
    sdelta[qb] = sd;
#pragma unroll
    for (int n = 0; n < DSTATE; ++n) Q[qb * DSTATE + n] = h[n];
}

__global__ void k_scan2(const float* __restrict__ sdelta, const float* __restrict__ Q,
                        const float* __restrict__ A_log, float* __restrict__ Hin)
{
    int g = blockIdx.x * 256 + threadIdx.x;
    if (g >= NB * DINNER) return;
    int b = g / DINNER, d = g - b * DINNER;
    float a[DSTATE], h[DSTATE];
#pragma unroll
    for (int n = 0; n < DSTATE; ++n) { a[n] = -__expf(A_log[d * DSTATE + n]); h[n] = 0.f; }
    for (int c = 0; c < NCHUNK; ++c) {
        size_t qb = ((size_t)(b * NCHUNK + c) * DINNER + d);
#pragma unroll
        for (int n = 0; n < DSTATE; ++n) Hin[qb * DSTATE + n] = h[n];
        float sd = sdelta[qb];
#pragma unroll
        for (int n = 0; n < DSTATE; ++n) h[n] = h[n] * __expf(a[n] * sd) + Q[qb * DSTATE + n];
    }
}

__global__ void k_scan3(const float* __restrict__ delta, const float* __restrict__ xc32,
                        const float* __restrict__ xdbl,  const float* __restrict__ A_log,
                        const float* __restrict__ Hin,   const float* __restrict__ ybuf,
                        const float* __restrict__ Dp,    const _Float16* __restrict__ z16,
                        _Float16* __restrict__ yact16)
{
    __shared__ float lC[CHUNK][DSTATE];
    int tmp = blockIdx.x;
    int dblk = tmp & 7; int c = (tmp >> 3) & 15; int b = tmp >> 7;
    int d0 = dblk * 256;

    for (int i = threadIdx.x; i < CHUNK * DSTATE; i += 256) {
        int t = i >> 4, j = i & 15;
        lC[t][j] = xdbl[((size_t)(b * SEQL + c * CHUNK + t)) * XDBLN + DTRANK + DSTATE + j];
    }
    __syncthreads();

    int d = d0 + threadIdx.x;
    float a[DSTATE], h0[DSTATE];
    size_t qb = ((size_t)(b * NCHUNK + c) * DINNER + d);
#pragma unroll
    for (int n = 0; n < DSTATE; ++n) {
        a[n]  = -__expf(A_log[d * DSTATE + n]);
        h0[n] = Hin[qb * DSTATE + n];
    }
    float Dd = Dp[d];
    float cum = 0.f;
    for (int t = 0; t < CHUNK; ++t) {
        size_t row = (size_t)b * SEQL + c * CHUNK + t;
        float de = delta[row * DINNER + d];
        cum += de;
        float corr = 0.f;
#pragma unroll
        for (int n = 0; n < DSTATE; ++n) corr += lC[t][n] * h0[n] * __expf(a[n] * cum);
        float y = ybuf[row * DINNER + d] + corr + xc32[row * DINNER + d] * Dd;
        float zv = (float)z16[row * DINNER + d];
        y *= zv / (1.f + __expf(-zv));                   // * silu(z)
        yact16[row * DINNER + d] = (_Float16)y;
    }
}

// =====================================================================
// host driver
// =====================================================================
extern "C" void kernel_launch(void* const* d_in, const int* in_sizes, int n_in,
                              void* d_out, int out_size, void* d_ws, size_t ws_size,
                              hipStream_t stream)
{
    const float* x = (const float*)d_in[0];
    float* out = (float*)d_out;

    char* w = (char*)d_ws; size_t off = 0;
    auto alloc = [&](size_t bytes) { void* p = w + off; off = (off + bytes + 255) & ~(size_t)255; return p; };
    _Float16* x16    = (_Float16*)alloc((size_t)MROWS * DMODEL * 2);
    _Float16* inw16  = (_Float16*)alloc((size_t)2 * DINNER * DMODEL * 2);
    _Float16* xpw16  = (_Float16*)alloc((size_t)XDBLN * DINNER * 2);
    _Float16* dtpw16 = (_Float16*)alloc((size_t)DINNER * DTRANK * 2);
    _Float16* outw16 = (_Float16*)alloc((size_t)DMODEL * DINNER * 2);
    _Float16* xi16   = (_Float16*)alloc((size_t)MROWS * DINNER * 2);
    _Float16* z16    = (_Float16*)alloc((size_t)MROWS * DINNER * 2);
    float*    xc32   = (float*)   alloc((size_t)MROWS * DINNER * 4);
    _Float16* xc16   = (_Float16*)alloc((size_t)MROWS * DINNER * 2);
    float*    xdbl   = (float*)   alloc((size_t)MROWS * XDBLN * 4);
    _Float16* dt16   = (_Float16*)alloc((size_t)MROWS * DTRANK * 2);
    float*    delta  = (float*)   alloc((size_t)MROWS * DINNER * 4);
    float*    ybuf   = (float*)   alloc((size_t)MROWS * DINNER * 4);
    _Float16* yact16 = (_Float16*)alloc((size_t)MROWS * DINNER * 2);
    float*    sdel   = (float*)   alloc((size_t)NB * NCHUNK * DINNER * 4);
    float*    Q      = (float*)   alloc((size_t)NB * NCHUNK * DINNER * DSTATE * 4);
    float*    Hin    = (float*)   alloc((size_t)NB * NCHUNK * DINNER * DSTATE * 4);

    for (int dir = 0; dir < 2; ++dir) {
        const float* in_w  = (const float*)d_in[1 + 9 * dir];
        const float* cw    = (const float*)d_in[2 + 9 * dir];
        const float* cb    = (const float*)d_in[3 + 9 * dir];
        const float* xp_w  = (const float*)d_in[4 + 9 * dir];
        const float* dtp_w = (const float*)d_in[5 + 9 * dir];
        const float* dtp_b = (const float*)d_in[6 + 9 * dir];
        const float* A_log = (const float*)d_in[7 + 9 * dir];
        const float* Dp    = (const float*)d_in[8 + 9 * dir];
        const float* out_w = (const float*)d_in[9 + 9 * dir];

        k_flip_cvt_x<<<(MROWS * DMODEL) / 256, 256, 0, stream>>>(x, x16, dir);
        k_cvt_f32_f16<<<(2 * DINNER * DMODEL) / 256, 256, 0, stream>>>(in_w, inw16, 2 * DINNER * DMODEL);
        k_cvt_f32_f16<<<(XDBLN * DINNER) / 256, 256, 0, stream>>>(xp_w, xpw16, XDBLN * DINNER);
        k_cvt_f32_f16<<<(DINNER * DTRANK) / 256, 256, 0, stream>>>(dtp_w, dtpw16, DINNER * DTRANK);
        k_cvt_f32_f16<<<(DMODEL * DINNER) / 256, 256, 0, stream>>>(out_w, outw16, DMODEL * DINNER);

        k_gemm_split<<<dim3((2 * DINNER) / 64, MROWS / 32), 256, 0, stream>>>(
            x16, inw16, xi16, z16, DMODEL, 2 * DINNER);
        k_conv_silu<<<(MROWS * DINNER) / 256, 256, 0, stream>>>(xi16, cw, cb, xc32, xc16);
        k_gemm_xdbl<<<dim3((XDBLN + 63) / 64, MROWS / 32), 256, 0, stream>>>(
            xc16, xpw16, xdbl, dt16, DINNER, XDBLN);
        k_gemm_delta<<<dim3(DINNER / 64, MROWS / 32), 256, 0, stream>>>(
            dt16, dtpw16, dtp_b, delta, DTRANK, DINNER);
        k_scan1<<<NB * NCHUNK * (DINNER / 256), 256, 0, stream>>>(
            delta, xc32, xdbl, A_log, ybuf, sdel, Q);
        k_scan2<<<(NB * DINNER) / 256, 256, 0, stream>>>(sdel, Q, A_log, Hin);
        k_scan3<<<NB * NCHUNK * (DINNER / 256), 256, 0, stream>>>(
            delta, xc32, xdbl, A_log, Hin, ybuf, Dp, z16, yact16);
        k_gemm_out<<<dim3(DMODEL / 64, MROWS / 32), 256, 0, stream>>>(
            yact16, outw16, out, DINNER, DMODEL, dir);
    }
}
